// PhysicsAttention_37460704756482
// MI455X (gfx1250) — matrix operand order repaired and checked by reading, not compiled, not run on hardware
//
#include <hip/hip_runtime.h>

// ---------------------------------------------------------------------------
// PhysicsAttention for MI455X (gfx1250), wave32 + v_wmma_f32_16x16x32_f16
// ---------------------------------------------------------------------------

typedef _Float16 h16v __attribute__((ext_vector_type(16)));
typedef _Float16 h8v  __attribute__((ext_vector_type(8)));
typedef float    f8v  __attribute__((ext_vector_type(8)));

union H16 { h16v v; h8v h[2]; _Float16 e[16]; };

#define WMMA_F16(A, B, C) \
  __builtin_amdgcn_wmma_f32_16x16x32_f16(false, (A), false, (B), (short)0, (C), false, false)

constexpr int B_ = 4, N_ = 32768, W_ = 256, M_ = 64, H_ = 8, D_ = 32;
constexpr int CHUNK = 64;                 // nodes per tile
constexpr int NBLK  = 64;                 // blocks per batch in kernel 1
constexpr int ITERS = N_ / (NBLK * CHUNK); // 8

// ---------------------------------------------------------------------------
// k0: convert Ws (f32) -> f16 scratch copy
// ---------------------------------------------------------------------------
__global__ void k0_cvt(const float* __restrict__ Ws, _Float16* __restrict__ Wsf) {
  int i = blockIdx.x * 256 + threadIdx.x;
  if (i < M_ * W_) Wsf[i] = (_Float16)Ws[i];
}

// ---------------------------------------------------------------------------
// k1: fused  logits(WMMA) -> softmax(in-register) -> pooling GEMM(WMMA)
//     grid (NBLK, B), 256 threads = 8 waves; each block loops 8 chunks of 64
//     nodes, keeping z partial sums resident in 64 f32 VGPRs per wave.
// ---------------------------------------------------------------------------
__global__ __launch_bounds__(256) void k1_main(
    const float* __restrict__ x, const _Float16* __restrict__ Wsf,
    const float* __restrict__ bs, _Float16* __restrict__ wg,
    float* __restrict__ zpart, float* __restrict__ wsumpart)
{
  __shared__ _Float16 xs[CHUNK][264];   // x chunk, node-major, f16 (33792 B)
  __shared__ _Float16 wsp[M_][72];      // Ws K-panel (9216 B)
  __shared__ _Float16 wT[M_][72];       // w transposed: [m][node] (9216 B)
  __shared__ float    wsumLoc[M_];

  const int tid    = threadIdx.x;
  const int lane   = tid & 31;
  const int wv     = tid >> 5;
  const int laneLo = lane & 15;
  const int laneHi = lane >> 4;
  const int b      = blockIdx.y;
  const int blk    = blockIdx.x;

  // persistent z accumulators: wave owns wcol-tiles {2wv,2wv+1} x m-tiles 0..3
  f8v zacc[4][2];
#pragma unroll
  for (int t = 0; t < 4; ++t)
#pragma unroll
    for (int u = 0; u < 2; ++u)
#pragma unroll
      for (int j = 0; j < 8; ++j) zacc[t][u][j] = 0.f;

  // per-lane slice-bias values matching logitsT C-fragment layout
  float bsv[4][8];
#pragma unroll
  for (int t = 0; t < 4; ++t)
#pragma unroll
    for (int i = 0; i < 8; ++i) bsv[t][i] = bs[t * 16 + laneHi * 8 + i];

  if (tid < M_) wsumLoc[tid] = 0.f;
  __syncthreads();

  for (int it = 0; it < ITERS; ++it) {
    const int node0 = blk * CHUNK + it * (NBLK * CHUNK);

    // ---- stage x chunk (coalesced f32 -> f16 LDS) ----
    for (int i = 0; i < (CHUNK * W_) / 256; ++i) {
      int idx = i * 256 + tid;
      int r = idx >> 8, c = idx & 255;
      xs[r][c] = (_Float16)x[((size_t)b * N_ + node0 + r) * (size_t)W_ + c];
    }
    __syncthreads();

    // ---- GEMM1: logitsT (m x node) on waves 0..3, K = W in 4 panels ----
    f8v lg[4];
#pragma unroll
    for (int t = 0; t < 4; ++t)
#pragma unroll
      for (int j = 0; j < 8; ++j) lg[t][j] = 0.f;

    for (int kp = 0; kp < 4; ++kp) {
      for (int i = 0; i < (M_ * 64) / 256; ++i) {
        int idx = i * 256 + tid;
        int r = idx >> 6, c = idx & 63;
        wsp[r][c] = Wsf[r * W_ + kp * 64 + c];
      }
      __syncthreads();
      if (wv < 4) {
#pragma unroll
        for (int ks = 0; ks < 2; ++ks) {
          H16 bf;                                    // B = x^T: col = node
          const int node = wv * 16 + laneLo;
          const int kb = kp * 64 + ks * 32 + laneHi * 16;
          bf.h[0] = *(const h8v*)&xs[node][kb];
          bf.h[1] = *(const h8v*)&xs[node][kb + 8];
#pragma unroll
          for (int t = 0; t < 4; ++t) {
            H16 af;                                  // A = Ws: row = m
            const int m = t * 16 + laneLo;
            const int ka = ks * 32 + laneHi * 16;
            af.h[0] = *(const h8v*)&wsp[m][ka];
            af.h[1] = *(const h8v*)&wsp[m][ka + 8];
            lg[t] = WMMA_F16(af.v, bf.v, lg[t]);
          }
        }
      }
      __syncthreads();
    }

    // ---- softmax over m (column of logitsT) in registers ----
    if (wv < 4) {
      float mx = -3.4e38f;
#pragma unroll
      for (int t = 0; t < 4; ++t)
#pragma unroll
        for (int i = 0; i < 8; ++i) {
          lg[t][i] += bsv[t][i];
          mx = fmaxf(mx, lg[t][i]);
        }
      mx = fmaxf(mx, __shfl_xor(mx, 16));
      float s = 0.f;
#pragma unroll
      for (int t = 0; t < 4; ++t)
#pragma unroll
        for (int i = 0; i < 8; ++i) {
          float e = __expf(lg[t][i] - mx);
          lg[t][i] = e;
          s += e;
        }
      s += __shfl_xor(s, 16);
      const float inv = 1.f / s;
      const int node = wv * 16 + laneLo;
#pragma unroll
      for (int t = 0; t < 4; ++t)
#pragma unroll
        for (int i = 0; i < 8; ++i)
          wT[t * 16 + laneHi * 8 + i][node] = (_Float16)(lg[t][i] * inv);
    }
    __syncthreads();

    // ---- wsum partial + coalesced global w (f16) store ----
    if (tid < M_) {
      float s = 0.f;
#pragma unroll 8
      for (int n = 0; n < CHUNK; ++n) s += (float)wT[tid][n];
      wsumLoc[tid] += s;
    }
    for (int i = 0; i < (CHUNK * M_) / 256; ++i) {
      int idx = i * 256 + tid;
      int nn = idx >> 6, m = idx & 63;
      wg[((size_t)b * N_ + node0 + nn) * (size_t)M_ + m] = wT[m][nn];
    }

    // ---- GEMM2: z += wT(m x node) @ x(node x W), all 8 waves ----
#pragma unroll
    for (int ks = 0; ks < 2; ++ks) {
      H16 af[4];
#pragma unroll
      for (int t = 0; t < 4; ++t) {
        const int m = t * 16 + laneLo;
        const int ka = ks * 32 + laneHi * 16;
        af[t].h[0] = *(const h8v*)&wT[m][ka];
        af[t].h[1] = *(const h8v*)&wT[m][ka + 8];
      }
#pragma unroll
      for (int u = 0; u < 2; ++u) {
        const int wcol = (2 * wv + u) * 16 + laneLo;
        const int nb = ks * 32 + laneHi * 16;
        H16 bf;                                      // transpose gather of x
#pragma unroll
        for (int e = 0; e < 16; ++e) bf.e[e] = xs[nb + e][wcol];
#pragma unroll
        for (int t = 0; t < 4; ++t) zacc[t][u] = WMMA_F16(af[t].v, bf.v, zacc[t][u]);
      }
    }
    __syncthreads();   // protect xs/wT before next chunk restage
  }

  // ---- deterministic per-block partial stores ----
#pragma unroll
  for (int t = 0; t < 4; ++t)
#pragma unroll
    for (int u = 0; u < 2; ++u)
#pragma unroll
      for (int i = 0; i < 8; ++i) {
        int m = t * 16 + laneHi * 8 + i;
        int wcol = (2 * wv + u) * 16 + laneLo;
        zpart[(((size_t)b * NBLK + blk) * M_ + m) * (size_t)W_ + wcol] = zacc[t][u][i];
      }
  if (tid < M_) wsumpart[((size_t)b * NBLK + blk) * M_ + tid] = wsumLoc[tid];
}

// ---------------------------------------------------------------------------
// k2a: reduce partials, z = sum(zpart)/max(sum(wsum),1e-8)   (deterministic)
// ---------------------------------------------------------------------------
__global__ void k2a_reduce(const float* __restrict__ zpart,
                           const float* __restrict__ wsumpart,
                           float* __restrict__ z)
{
  int i = blockIdx.x * 256 + threadIdx.x;     // 0 .. B*M*W-1
  int b = i >> 14;
  int m = (i >> 8) & 63;
  int local = i & 16383;
  float s = 0.f, ws = 0.f;
  for (int blk = 0; blk < NBLK; ++blk) {
    s  += zpart[((size_t)(b * NBLK + blk)) * 16384 + local];
    ws += wsumpart[(b * NBLK + blk) * M_ + m];
  }
  z[i] = s / fmaxf(ws, 1e-8f);
}

// ---------------------------------------------------------------------------
// k2b: qkv = z @ Wqkv^T + bqkv     grid (12, B)
// ---------------------------------------------------------------------------
__global__ __launch_bounds__(256) void k2b_qkv(
    const float* __restrict__ z, const float* __restrict__ Wqkv,
    const float* __restrict__ bqkv, float* __restrict__ qkv)
{
  __shared__ _Float16 zsh[M_][264];
  const int b = blockIdx.y;
  const int c0 = blockIdx.x * 64;
  const int tid = threadIdx.x;
  for (int i = 0; i < 64; ++i) {
    int idx = i * 256 + tid;
    int r = idx >> 8, c = idx & 255;
    zsh[r][c] = (_Float16)z[((size_t)b * M_ + r) * W_ + c];
  }
  __syncthreads();
  const int c = c0 + (tid & 63);
  const int mg = tid >> 6;
  float acc[16];
#pragma unroll
  for (int j = 0; j < 16; ++j) acc[j] = 0.f;
  for (int k = 0; k < W_; ++k) {
    float wq = Wqkv[(size_t)c * W_ + k];
#pragma unroll
    for (int j = 0; j < 16; ++j) acc[j] += (float)zsh[mg * 16 + j][k] * wq;
  }
  const float bq = bqkv[c];
#pragma unroll
  for (int j = 0; j < 16; ++j)
    qkv[((size_t)b * M_ + (mg * 16 + j)) * 768 + c] = acc[j] + bq;
}

// ---------------------------------------------------------------------------
// k2c: per-(b,h) 64-token attention, f32 VALU        grid (H, B)
// ---------------------------------------------------------------------------
__global__ __launch_bounds__(256) void k2c_attn(const float* __restrict__ qkv,
                                                float* __restrict__ obuf)
{
  __shared__ float qs[M_][D_ + 1], ksh[M_][D_ + 1], vsh[M_][D_ + 1];
  __shared__ float sc[M_][M_ + 1];
  const int b = blockIdx.y, h = blockIdx.x;
  const int tid = threadIdx.x;
  for (int i = 0; i < 8; ++i) {
    int idx = i * 256 + tid;
    int m = idx >> 5, d = idx & 31;
    size_t base = ((size_t)b * M_ + m) * 768 + h * D_ + d;
    qs[m][d]  = qkv[base];
    ksh[m][d] = qkv[base + 256];
    vsh[m][d] = qkv[base + 512];
  }
  __syncthreads();
  for (int i = 0; i < 16; ++i) {
    int idx = i * 256 + tid;
    int qm = idx >> 6, km = idx & 63;
    float s = 0.f;
#pragma unroll
    for (int d = 0; d < D_; ++d) s += qs[qm][d] * ksh[km][d];
    sc[qm][km] = s * 0.17677669529663687f;   // 1/sqrt(32)
  }
  __syncthreads();
  if (tid < M_) {
    float mx = -3.4e38f;
    for (int k = 0; k < M_; ++k) mx = fmaxf(mx, sc[tid][k]);
    float s = 0.f;
    for (int k = 0; k < M_; ++k) { float e = __expf(sc[tid][k] - mx); sc[tid][k] = e; s += e; }
    const float inv = 1.f / s;
    for (int k = 0; k < M_; ++k) sc[tid][k] *= inv;
  }
  __syncthreads();
  for (int i = 0; i < 8; ++i) {
    int idx = i * 256 + tid;
    int m = idx >> 5, d = idx & 31;
    float s = 0.f;
    for (int k = 0; k < M_; ++k) s += sc[m][k] * vsh[k][d];
    obuf[((size_t)b * M_ + m) * W_ + h * D_ + d] = s;
  }
}

// ---------------------------------------------------------------------------
// k2d: z_out = o @ Wo^T + bo, stored transposed as f16 zoT[b][w][m]  grid(4,B)
// ---------------------------------------------------------------------------
__global__ __launch_bounds__(256) void k2d_out(
    const float* __restrict__ obuf, const float* __restrict__ Wo,
    const float* __restrict__ bo, _Float16* __restrict__ zoT)
{
  __shared__ _Float16 osh[M_][264];
  const int b = blockIdx.y;
  const int c0 = blockIdx.x * 64;
  const int tid = threadIdx.x;
  for (int i = 0; i < 64; ++i) {
    int idx = i * 256 + tid;
    int r = idx >> 8, c = idx & 255;
    osh[r][c] = (_Float16)obuf[((size_t)b * M_ + r) * W_ + c];
  }
  __syncthreads();
  const int c = c0 + (tid & 63);
  const int mg = tid >> 6;
  float acc[16];
#pragma unroll
  for (int j = 0; j < 16; ++j) acc[j] = 0.f;
  for (int k = 0; k < W_; ++k) {
    float wv2 = Wo[(size_t)c * W_ + k];
#pragma unroll
    for (int j = 0; j < 16; ++j) acc[j] += (float)osh[mg * 16 + j][k] * wv2;
  }
  const float bb = bo[c];
#pragma unroll
  for (int j = 0; j < 16; ++j)
    zoT[((size_t)b * W_ + c) * M_ + (mg * 16 + j)] = (_Float16)(acc[j] + bb);
}

// ---------------------------------------------------------------------------
// k3: out = w @ z_out  (WMMA broadcast back to nodes)    grid (N/64, B)
// ---------------------------------------------------------------------------
__global__ __launch_bounds__(256) void k3_bcast(
    const _Float16* __restrict__ wg, const _Float16* __restrict__ zoT,
    float* __restrict__ out)
{
  __shared__ _Float16 wsh[CHUNK][72];    // [node][m]
  __shared__ _Float16 zTsh[W_][72];      // [wcol][m]
  const int b = blockIdx.y;
  const int node0 = blockIdx.x * CHUNK;
  const int tid = threadIdx.x;
  const int lane = tid & 31, wv = tid >> 5;
  const int laneLo = lane & 15, laneHi = lane >> 4;

  for (int i = 0; i < 16; ++i) {
    int idx = i * 256 + tid;
    int nn = idx >> 6, m = idx & 63;
    wsh[nn][m] = wg[((size_t)b * N_ + node0 + nn) * (size_t)M_ + m];
  }
  for (int i = 0; i < 64; ++i) {
    int idx = i * 256 + tid;
    int c = idx >> 6, m = idx & 63;
    zTsh[c][m] = zoT[((size_t)b * W_ + c) * M_ + m];
  }
  __syncthreads();

  f8v acc[4][2];
#pragma unroll
  for (int t = 0; t < 4; ++t)
#pragma unroll
    for (int u = 0; u < 2; ++u)
#pragma unroll
      for (int j = 0; j < 8; ++j) acc[t][u][j] = 0.f;

#pragma unroll
  for (int ks = 0; ks < 2; ++ks) {
    H16 af[4];
#pragma unroll
    for (int t = 0; t < 4; ++t) {                 // A = w: row = node, K = m
      const int nn = t * 16 + laneLo;
      const int ka = ks * 32 + laneHi * 16;
      af[t].h[0] = *(const h8v*)&wsh[nn][ka];
      af[t].h[1] = *(const h8v*)&wsh[nn][ka + 8];
    }
#pragma unroll
    for (int u = 0; u < 2; ++u) {                 // B = z_out: col = wcol
      const int c = (2 * wv + u) * 16 + laneLo;
      const int kb = ks * 32 + laneHi * 16;
      H16 bf;
      bf.h[0] = *(const h8v*)&zTsh[c][kb];
      bf.h[1] = *(const h8v*)&zTsh[c][kb + 8];
#pragma unroll
      for (int t = 0; t < 4; ++t) acc[t][u] = WMMA_F16(af[t].v, bf.v, acc[t][u]);
    }
  }

#pragma unroll
  for (int t = 0; t < 4; ++t)
#pragma unroll
    for (int u = 0; u < 2; ++u)
#pragma unroll
      for (int i = 0; i < 8; ++i) {
        int nn = node0 + t * 16 + laneHi * 8 + i;
        int c = (2 * wv + u) * 16 + laneLo;
        out[((size_t)b * N_ + nn) * (size_t)W_ + c] = acc[t][u][i];
      }
}

// ---------------------------------------------------------------------------
extern "C" void kernel_launch(void* const* d_in, const int* in_sizes, int n_in,
                              void* d_out, int out_size, void* d_ws, size_t ws_size,
                              hipStream_t stream)
{
  (void)in_sizes; (void)n_in; (void)out_size; (void)ws_size;

  const float* x    = (const float*)d_in[0];
  const float* Ws   = (const float*)d_in[1];
  const float* bs   = (const float*)d_in[2];
  const float* Wqkv = (const float*)d_in[3];
  const float* bqkv = (const float*)d_in[4];
  const float* Wo   = (const float*)d_in[5];
  const float* bo   = (const float*)d_in[6];
  float* out = (float*)d_out;

  char* ws = (char*)d_ws;
  constexpr size_t o_Wsf   = 0;
  constexpr size_t o_zpart = o_Wsf   + (size_t)M_ * W_ * 2;             //  32 KB
  constexpr size_t o_wsump = o_zpart + (size_t)B_ * NBLK * M_ * W_ * 4; // +16 MB
  constexpr size_t o_w     = o_wsump + (size_t)B_ * NBLK * M_ * 4;      // +64 KB
  constexpr size_t o_z     = o_w     + (size_t)B_ * N_ * M_ * 2;        // +16 MB
  constexpr size_t o_qkv   = o_z     + (size_t)B_ * M_ * W_ * 4;
  constexpr size_t o_obuf  = o_qkv   + (size_t)B_ * M_ * 768 * 4;
  constexpr size_t o_zoT   = o_obuf  + (size_t)B_ * M_ * W_ * 4;

  _Float16* Wsf  = (_Float16*)(ws + o_Wsf);
  float*    zpart = (float*)(ws + o_zpart);
  float*    wsump = (float*)(ws + o_wsump);
  _Float16* wg   = (_Float16*)(ws + o_w);
  float*    z    = (float*)(ws + o_z);
  float*    qkv  = (float*)(ws + o_qkv);
  float*    obuf = (float*)(ws + o_obuf);
  _Float16* zoT  = (_Float16*)(ws + o_zoT);

  k0_cvt    <<<64, 256, 0, stream>>>(Ws, Wsf);
  k1_main   <<<dim3(NBLK, B_), 256, 0, stream>>>(x, Wsf, bs, wg, zpart, wsump);
  k2a_reduce<<<(B_ * M_ * W_) / 256, 256, 0, stream>>>(zpart, wsump, z);
  k2b_qkv   <<<dim3(12, B_), 256, 0, stream>>>(z, Wqkv, bqkv, qkv);
  k2c_attn  <<<dim3(H_, B_), 256, 0, stream>>>(qkv, obuf);
  k2d_out   <<<dim3(4, B_), 256, 0, stream>>>(obuf, Wo, bo, zoT);
  k3_bcast  <<<dim3(N_ / CHUNK, B_), 256, 0, stream>>>(wg, zoT, out);
}
